// Gemma4Experts_62294205661276
// MI455X (gfx1250) — compile-verified, matrix operand order checked by reading, not run on hardware
//
#include <hip/hip_runtime.h>

// MoE grouped FFN:  out = (gelu_tanh(x@w1[e]^T) * (x@w3[e]^T)) @ w2[e]^T
// E=8, T=16384, D=2048, H=1024.  Compute-bound -> bf16 WMMA
// (V_WMMA_F32_16X16X32_BF16) with f32 accumulation.
//
// Phase 0: one streaming pass converts x/w1/w3/w2 to bf16 in workspace
//          (~0.5 GB @ 23.3 TB/s; makes GEMM1's working set L2-resident and
//          strips all cvt work out of the GEMM hot loops).
// Phase 1: GEMM1 fuses both x@w1^T and x@w3^T (shared A tile) + gelu-gate,
//          h stored bf16.
// Phase 2: GEMM2 h@w2^T -> f32 out.
// GEMMs: 128x64 block tile, 8 wave32s, double-buffered LDS (1 barrier per
// 32-deep k-step), raw uint4 global->LDS copies, ds_load_b128 fragments.

#define TDIM 16384
#define DDIM 2048
#define HDIM 1024
#define NEXP 8

typedef __bf16 bf16_t;
typedef __bf16 v16bf __attribute__((ext_vector_type(16)));
typedef __bf16 v2bf  __attribute__((ext_vector_type(2)));
typedef float  v8f   __attribute__((ext_vector_type(8)));
typedef float  v2f   __attribute__((ext_vector_type(2)));

// LDS row pitch for a 32-element bf16 row: 40 elems = 80 bytes.
// 80B stride = 20-bank step -> 16 rows hit 16 distinct bank quads, so the
// 16-lane ds_load_b128 fragment fetch is conflict-free.  Row-internal uint4
// stores land at 0/16/32/48B -> 16B aligned.
#define LDS_PITCH 40

// Vector fptrunc -> native V_CVT_PK_BF16_F32 (verified in round-3 disasm).
__device__ __forceinline__ unsigned pack_bf16(float a, float b) {
  v2f  s = {a, b};
  v2bf d = __builtin_convertvector(s, v2bf);
  return __builtin_bit_cast(unsigned, d);
}

union FragU { v16bf v; uint4 u[2]; };

// WMMA 16-bit A/B fragment (16 rows x 32 K) from an LDS tile whose rows are
// [row][32 bf16] at LDS_PITCH.  Per ISA 7.12.2: lanes 0-15 hold row=lane with
// K 0..7 (v0..3) and K 16..23 (v4..7); lanes 16-31 hold K 8..15 / 24..31.
__device__ __forceinline__ v16bf load_frag(const bf16_t* tile, int lane) {
  const int row = lane & 15;
  const int sel = (lane >> 4) & 1;
  const bf16_t* p = tile + row * LDS_PITCH + sel * 8;
  FragU f;
  f.u[0] = *(const uint4*)(p);
  f.u[1] = *(const uint4*)(p + 16);
  return f.v;
}

// Branch-free tanh: 1 - 2/(exp(2u)+1), clamped so v_exp_f32 never overflows.
__device__ __forceinline__ float tanh_fast(float u) {
  float cu = fminf(fmaxf(u, -8.0f), 8.0f);
  float e  = __expf(2.0f * cu);
  return 1.0f - 2.0f * __builtin_amdgcn_rcpf(e + 1.0f);
}
__device__ __forceinline__ float gelu_tanh(float x) {
  float u = 0.7978845608028654f * (x + 0.044715f * x * x * x);
  return 0.5f * x * (1.0f + tanh_fast(u));
}

// Map a flat 128-row tile index onto (expert, global row start, valid rows).
__device__ __forceinline__ bool locate_tile(const int* __restrict__ counts, int rtile,
                                            int& expert, int& rowStart, int& rowsValid) {
  int off = 0, acc = 0;
#pragma unroll
  for (int e = 0; e < NEXP; ++e) {
    int c  = counts[e];
    int nt = (c + 127) >> 7;
    if (rtile < acc + nt) {
      int local = (rtile - acc) << 7;
      expert    = e;
      rowStart  = off + local;
      rowsValid = c - local;
      if (rowsValid > 128) rowsValid = 128;
      return true;
    }
    acc += nt;
    off += c;
  }
  return false;
}

// ---------------- Phase 0: f32 -> bf16 streaming conversion ------------------
__global__ __launch_bounds__(256)
void cvt_bf16_kernel(const float* __restrict__ src,
                     unsigned short* __restrict__ dst, int n4) {
  int i = blockIdx.x * 256 + threadIdx.x;
  if (i < n4) {
    float4 v = ((const float4*)src)[i];
    uint2 p;
    p.x = pack_bf16(v.x, v.y);
    p.y = pack_bf16(v.z, v.w);
    ((uint2*)dst)[i] = p;
  }
}

// ---------------- GEMM1: h = gelu(x@w1^T) * (x@w3^T), bf16 in/out ------------

struct G1Regs { uint4 a[2]; uint4 b[2]; };

__device__ __forceinline__ void g1_load(G1Regs& r,
                                        const unsigned short* __restrict__ xb,
                                        const unsigned short* __restrict__ wb1,
                                        const unsigned short* __restrict__ wb3,
                                        int rowStart, int colBase, int k0, int tid) {
#pragma unroll
  for (int j = 0; j < 2; ++j) {
    int i   = tid + 256 * j;                  // A: 128 rows x 4 uint4
    int row = i >> 2;
    int kq  = (i & 3) << 3;
    int gr  = rowStart + row;
    gr = gr < TDIM ? gr : TDIM - 1;           // clamp tail tile
    r.a[j] = *(const uint4*)(xb + (size_t)gr * DDIM + k0 + kq);

    int m    = i >> 8;                        // 0 -> w1 tile, 1 -> w3 tile
    int ii   = i & 255;                       // 64 rows x 4 uint4
    int brow = ii >> 2;
    int bkq  = (ii & 3) << 3;
    const unsigned short* wsrc = m ? wb3 : wb1;
    r.b[j] = *(const uint4*)(wsrc + (size_t)(colBase + brow) * DDIM + k0 + bkq);
  }
}

__device__ __forceinline__ void g1_store(const G1Regs& r,
                                         bf16_t* sA, bf16_t* sB1, bf16_t* sB3, int tid) {
#pragma unroll
  for (int j = 0; j < 2; ++j) {
    int i   = tid + 256 * j;
    int row = i >> 2;
    int kq  = (i & 3) << 3;
    *(uint4*)&sA[row * LDS_PITCH + kq] = r.a[j];

    int m    = i >> 8;
    int ii   = i & 255;
    int brow = ii >> 2;
    int bkq  = (ii & 3) << 3;
    bf16_t* dst = m ? sB3 : sB1;
    *(uint4*)&dst[brow * LDS_PITCH + bkq] = r.b[j];
  }
}

__global__ __launch_bounds__(256)
void moe_gemm1_kernel(const unsigned short* __restrict__ xb,
                      const unsigned short* __restrict__ w1b,
                      const unsigned short* __restrict__ w3b,
                      const int* __restrict__ counts,
                      unsigned short* __restrict__ h) {
  __shared__ __align__(16) bf16_t sA [2][128 * LDS_PITCH];
  __shared__ __align__(16) bf16_t sB1[2][ 64 * LDS_PITCH];
  __shared__ __align__(16) bf16_t sB3[2][ 64 * LDS_PITCH];

  int expert, rowStart, rowsValid;
  if (!locate_tile(counts, blockIdx.y, expert, rowStart, rowsValid)) return;

  const int colBase = blockIdx.x * 64;          // H columns covered: 64
  const int tid  = threadIdx.x;
  const int lane = tid & 31;
  const int wave = tid >> 5;                    // 8 waves, 16 rows each

  const unsigned short* wb1 = w1b + (size_t)expert * HDIM * DDIM;
  const unsigned short* wb3 = w3b + (size_t)expert * HDIM * DDIM;

  v8f zero = {0.f, 0.f, 0.f, 0.f, 0.f, 0.f, 0.f, 0.f};
  // acc[0..3] = x@w1^T tiles, acc[4..7] = x@w3^T tiles
  v8f acc[8];
#pragma unroll
  for (int n = 0; n < 8; ++n) acc[n] = zero;

  const int NK = DDIM / 32;
  G1Regs rg;
  g1_load(rg, xb, wb1, wb3, rowStart, colBase, 0, tid);
  g1_store(rg, sA[0], sB1[0], sB3[0], tid);
  __syncthreads();

  for (int kt = 0; kt < NK; ++kt) {
    const int cur = kt & 1, nxt = cur ^ 1;
    if (kt + 1 < NK)
      g1_load(rg, xb, wb1, wb3, rowStart, colBase, (kt + 1) * 32, tid);

    const bf16_t* b1base = sB1[cur];
    const bf16_t* b3base = sB3[cur];
    v16bf afrag = load_frag(&sA[cur][wave * 16 * LDS_PITCH], lane);

    // 1-deep software pipeline over the 8 B tiles.
    v16bf bcur = load_frag(b1base, lane);
#pragma unroll
    for (int n = 0; n < 8; ++n) {
      v16bf bnext;
      if (n < 7) {
        int m = n + 1;
        const bf16_t* tp = (m < 4) ? (b1base + m * 16 * LDS_PITCH)
                                   : (b3base + (m - 4) * 16 * LDS_PITCH);
        bnext = load_frag(tp, lane);
      }
      acc[n] = __builtin_amdgcn_wmma_f32_16x16x32_bf16(
          false, afrag, false, bcur, (short)0, acc[n], false, false);
      if (n < 7) bcur = bnext;
    }

    if (kt + 1 < NK)
      g1_store(rg, sA[nxt], sB1[nxt], sB3[nxt], tid);
    __syncthreads();
  }

  // ---- epilogue: gate activation, bf16 store of h
  // C/D layout: lane L, vgpr r -> m = r + 8*(L>=16), n = L&15
  const int nlane = lane & 15;
  const int mhi   = (lane >> 4) << 3;
#pragma unroll
  for (int n = 0; n < 4; ++n) {
    int col = colBase + n * 16 + nlane;
#pragma unroll
    for (int r = 0; r < 8; ++r) {
      int rowInBlock = wave * 16 + mhi + r;
      if (rowInBlock < rowsValid) {
        float g = gelu_tanh(acc[n][r]) * acc[n + 4][r];
        h[(size_t)(rowStart + rowInBlock) * HDIM + col] =
            (unsigned short)pack_bf16(g, g);
      }
    }
  }
}

// ---------------- GEMM2: out = h @ w2^T  (f32 output) ------------------------

struct G2Regs { uint4 a[2]; uint4 b; };

__device__ __forceinline__ void g2_load(G2Regs& r,
                                        const unsigned short* __restrict__ h,
                                        const unsigned short* __restrict__ wb2,
                                        int rowStart, int colBase, int k0, int tid) {
#pragma unroll
  for (int j = 0; j < 2; ++j) {
    int i   = tid + 256 * j;                  // A: 128 rows x 4 uint4
    int row = i >> 2;
    int kq  = (i & 3) << 3;
    int gr  = rowStart + row;
    gr = gr < TDIM ? gr : TDIM - 1;
    r.a[j] = *(const uint4*)(h + (size_t)gr * HDIM + k0 + kq);
  }
  {
    int brow = tid >> 2;                      // B: 64 rows x 4 uint4
    int bkq  = (tid & 3) << 3;
    r.b = *(const uint4*)(wb2 + (size_t)(colBase + brow) * HDIM + k0 + bkq);
  }
}

__device__ __forceinline__ void g2_store(const G2Regs& r, bf16_t* sA, bf16_t* sB, int tid) {
#pragma unroll
  for (int j = 0; j < 2; ++j) {
    int i   = tid + 256 * j;
    int row = i >> 2;
    int kq  = (i & 3) << 3;
    *(uint4*)&sA[row * LDS_PITCH + kq] = r.a[j];
  }
  {
    int brow = tid >> 2;
    int bkq  = (tid & 3) << 3;
    *(uint4*)&sB[brow * LDS_PITCH + bkq] = r.b;
  }
}

__global__ __launch_bounds__(256)
void moe_gemm2_kernel(const unsigned short* __restrict__ h,
                      const unsigned short* __restrict__ w2b,
                      const int* __restrict__ counts,
                      float* __restrict__ out) {
  __shared__ __align__(16) bf16_t sA[2][128 * LDS_PITCH];
  __shared__ __align__(16) bf16_t sB[2][ 64 * LDS_PITCH];

  int expert, rowStart, rowsValid;
  if (!locate_tile(counts, blockIdx.y, expert, rowStart, rowsValid)) return;

  const int colBase = blockIdx.x * 64;          // D columns covered: 64
  const int tid  = threadIdx.x;
  const int lane = tid & 31;
  const int wave = tid >> 5;

  const unsigned short* wb2 = w2b + (size_t)expert * DDIM * HDIM;

  v8f zero = {0.f, 0.f, 0.f, 0.f, 0.f, 0.f, 0.f, 0.f};
  v8f acc[4];
#pragma unroll
  for (int n = 0; n < 4; ++n) acc[n] = zero;

  const int NK = HDIM / 32;
  G2Regs rg;
  g2_load(rg, h, wb2, rowStart, colBase, 0, tid);
  g2_store(rg, sA[0], sB[0], tid);
  __syncthreads();

  for (int kt = 0; kt < NK; ++kt) {
    const int cur = kt & 1, nxt = cur ^ 1;
    if (kt + 1 < NK)
      g2_load(rg, h, wb2, rowStart, colBase, (kt + 1) * 32, tid);

    const bf16_t* bbase = sB[cur];
    v16bf afrag = load_frag(&sA[cur][wave * 16 * LDS_PITCH], lane);

    v16bf bcur = load_frag(bbase, lane);
#pragma unroll
    for (int n = 0; n < 4; ++n) {
      v16bf bnext;
      if (n < 3) bnext = load_frag(bbase + (n + 1) * 16 * LDS_PITCH, lane);
      acc[n] = __builtin_amdgcn_wmma_f32_16x16x32_bf16(
          false, afrag, false, bcur, (short)0, acc[n], false, false);
      if (n < 3) bcur = bnext;
    }

    if (kt + 1 < NK)
      g2_store(rg, sA[nxt], sB[nxt], tid);
    __syncthreads();
  }

  const int nlane = lane & 15;
  const int mhi   = (lane >> 4) << 3;
#pragma unroll
  for (int n = 0; n < 4; ++n) {
    int col = colBase + n * 16 + nlane;
#pragma unroll
    for (int r = 0; r < 8; ++r) {
      int rowInBlock = wave * 16 + mhi + r;
      if (rowInBlock < rowsValid) {
        out[(size_t)(rowStart + rowInBlock) * DDIM + col] = acc[n][r];
      }
    }
  }
}

extern "C" void kernel_launch(void* const* d_in, const int* in_sizes, int n_in,
                              void* d_out, int out_size, void* d_ws, size_t ws_size,
                              hipStream_t stream) {
  const float* x   = (const float*)d_in[0];
  const float* w1  = (const float*)d_in[1];
  const float* w2  = (const float*)d_in[2];
  const float* w3  = (const float*)d_in[3];
  const int*   cnt = (const int*)d_in[4];
  float* out = (float*)d_out;

  // Workspace layout (bf16 elements): xb | w1b | w3b | w2b | h   (~201 MB)
  unsigned short* ws  = (unsigned short*)d_ws;
  const size_t nX = (size_t)TDIM * DDIM;          // 33.55M
  const size_t nW = (size_t)NEXP * HDIM * DDIM;   // 16.78M (same for w1/w3/w2)
  unsigned short* xb  = ws;
  unsigned short* w1b = xb  + nX;
  unsigned short* w3b = w1b + nW;
  unsigned short* w2b = w3b + nW;
  unsigned short* hb  = w2b + nW;                 // [T, H] bf16

  dim3 blk(256, 1, 1);

  // Phase 0: f32 -> bf16 streaming conversion (float4 / uint2 vectorized).
  {
    int n4x = (int)(nX / 4), n4w = (int)(nW / 4);
    cvt_bf16_kernel<<<dim3((n4x + 255) / 256), blk, 0, stream>>>(x,  xb,  n4x);
    cvt_bf16_kernel<<<dim3((n4w + 255) / 256), blk, 0, stream>>>(w1, w1b, n4w);
    cvt_bf16_kernel<<<dim3((n4w + 255) / 256), blk, 0, stream>>>(w3, w3b, n4w);
    cvt_bf16_kernel<<<dim3((n4w + 255) / 256), blk, 0, stream>>>(w2, w2b, n4w);
  }

  const int maxRowTiles = TDIM / 128 + NEXP;      // ragged worst case = 136
  dim3 g1(HDIM / 64, maxRowTiles, 1);
  dim3 g2(DDIM / 64, maxRowTiles, 1);
  moe_gemm1_kernel<<<g1, blk, 0, stream>>>(xb, w1b, w3b, cnt, hb);
  moe_gemm2_kernel<<<g2, blk, 0, stream>>>(hb, w2b, cnt, out);
}
